// NaviNetwork_35338990911904
// MI455X (gfx1250) — matrix operand order repaired and checked by reading, not compiled, not run on hardware
//
#include <hip/hip_runtime.h>
#include <hip/hip_bf16.h>
#include <math.h>

// ---------------------------------------------------------------------------
// CDNA5 (gfx1250) transformer encoder-decoder.
// All GEMM-class work goes through v_wmma_f32_16x16x32_bf16 (wave32).
// fp32 activations/weights are converted to bf16 at the tile level; all
// accumulation, softmax, layernorm, gelu in fp32.
// ---------------------------------------------------------------------------

typedef __attribute__((ext_vector_type(16))) __bf16 bf16x16;
typedef __attribute__((ext_vector_type(8)))  __bf16 bf16x8;
typedef __attribute__((ext_vector_type(8)))  float  f32x8;

#define WMMA_BF16(a, b, c) \
  __builtin_amdgcn_wmma_f32_16x16x32_bf16(false, (a), false, (b), (short)0, (c), false, false)

__device__ __forceinline__ float gelu_exact(float x) {
  return 0.5f * x * (1.0f + erff(x * 0.7071067811865475f));
}

// ---------------------------------------------------------------------------
// GEMM: C[M,N] = act(A[M,K] @ W[K,N] + bias)
// N, K compile-time (1024 / 4096) so the W-tile gather uses immediate
// instruction offsets (one s_clause, no per-load 64-bit address adds).
// Block tile 128x128, 8 waves (4 M x 2 N), each wave a 32x64 tile =
// 2x4 WMMA frags (8 v_wmma per K-step per wave, 64 per block).
// Staging: straight-line, vectorized. A: 4x float4 -> one 32B bf16x16 LDS
// store per thread. W: lane owns column n, gathers 16 k-values (coalesced
// across lanes) -> transposed 32B bf16x16 LDS store. Fragments load as
// contiguous ds_load_b128 with k = 16*kh + s in both A(16x32) and B(32x16)
// fragments (same k order in both).  act: 0 = none, 1 = exact GELU
// ---------------------------------------------------------------------------
#define BM 128
#define BN 128
#define BK 32

template <int N, int K>
__global__ __launch_bounds__(256) void gemm_bf16_kernel(
    const float* __restrict__ A, const float* __restrict__ W,
    const float* __restrict__ bias, float* __restrict__ C, int act) {
  __shared__ __align__(64) __bf16 As[BM][BK];
  __shared__ __align__(64) __bf16 Bs[BN][BK];  // transposed: [n][k]

  const int tid   = threadIdx.x;
  const int lane  = tid & 31;
  const int wave  = tid >> 5;
  const int waveM = wave >> 1;   // 0..3
  const int waveN = wave & 1;    // 0..1
  const int m0    = lane & 15;
  const int kh    = lane >> 4;   // lane-half select

  const int blockM = blockIdx.y * BM;
  const int blockN = blockIdx.x * BN;

  // staging assignments (256 threads)
  const int arow = tid >> 1;          // 0..127
  const int aseg = (tid & 1) * 16;    // 0 or 16 (k offset)
  const int bn   = tid & 127;         // 0..127 (n column)
  const int bk0  = (tid >> 7) * 16;   // 0 or 16 (k offset)

  const float* Arow = A + (size_t)(blockM + arow) * K + aseg;   // + k0
  const float* Wcol = W + (size_t)bk0 * N + blockN + bn;        // + k0*N, e*N

  f32x8 acc[2][4];
#pragma unroll
  for (int i = 0; i < 2; i++)
#pragma unroll
    for (int j = 0; j < 4; j++) acc[i][j] = {};

  for (int k0 = 0; k0 < K; k0 += BK) {
    // ---- stage A tile: 16 consecutive fp32 -> bf16x16 (64B-aligned loads)
    {
      const float4* ap4 = (const float4*)(Arow + k0);
      float4 f0 = ap4[0], f1 = ap4[1], f2 = ap4[2], f3 = ap4[3];
      bf16x16 av;
      av[0]  = (__bf16)f0.x; av[1]  = (__bf16)f0.y; av[2]  = (__bf16)f0.z; av[3]  = (__bf16)f0.w;
      av[4]  = (__bf16)f1.x; av[5]  = (__bf16)f1.y; av[6]  = (__bf16)f1.z; av[7]  = (__bf16)f1.w;
      av[8]  = (__bf16)f2.x; av[9]  = (__bf16)f2.y; av[10] = (__bf16)f2.z; av[11] = (__bf16)f2.w;
      av[12] = (__bf16)f3.x; av[13] = (__bf16)f3.y; av[14] = (__bf16)f3.z; av[15] = (__bf16)f3.w;
      *(bf16x16*)&As[arow][aseg] = av;
    }
    // ---- stage W tile transposed: 16 k-gather loads, coalesced across
    //      lanes; constant stride N -> immediate offsets, one clause
    {
      const float* wp = Wcol + (size_t)k0 * N;
      bf16x16 bv;
#pragma unroll
      for (int e = 0; e < 16; e++) bv[e] = (__bf16)wp[e * N];
      *(bf16x16*)&Bs[bn][bk0] = bv;
    }
    __syncthreads();

    // ---- prefetch next K-step tiles into cache (global_prefetch_b8)
    if (k0 + BK < K) {
      __builtin_prefetch(Arow + k0 + BK, 0, 1);
      __builtin_prefetch(Wcol + (size_t)(k0 + BK) * N, 0, 1);
    }

    // ---- fragments + 8 WMMA
    bf16x16 afrag[2], bfrag[4];
#pragma unroll
    for (int i = 0; i < 2; i++) {
      const __bf16* ap = &As[waveM * 32 + i * 16 + m0][0];
      afrag[i] = *(const bf16x16*)(ap + kh * 16);      // K = kh*16 .. +15
    }
#pragma unroll
    for (int j = 0; j < 4; j++) {
      const __bf16* bp = &Bs[waveN * 64 + j * 16 + m0][0];
      bfrag[j] = *(const bf16x16*)(bp + kh * 16);      // K = kh*16 .. +15
    }
#pragma unroll
    for (int i = 0; i < 2; i++)
#pragma unroll
      for (int j = 0; j < 4; j++)
        acc[i][j] = WMMA_BF16(afrag[i], bfrag[j], acc[i][j]);
    __syncthreads();
  }

  // ---- epilogue: C 16x16 f32 layout -> row = base + r + 8*kh, col = m0
#pragma unroll
  for (int i = 0; i < 2; i++)
#pragma unroll
    for (int j = 0; j < 4; j++) {
      int col = blockN + waveN * 64 + j * 16 + m0;
      float bv = bias ? bias[col] : 0.0f;
#pragma unroll
      for (int r = 0; r < 8; r++) {
        int row = blockM + waveM * 32 + i * 16 + r + kh * 8;
        float v = acc[i][j][r] + bv;
        if (act == 1) v = gelu_exact(v);
        C[(size_t)row * N + col] = v;
      }
    }
}

// ---------------------------------------------------------------------------
// Attention: one wave per (b, h, 16 query rows). Single pass (reference
// softmax = exp(clip(s,+-50)) with safe-divide; no running max needed).
// Per 32-key tile: S = Q K^T (4 WMMA), exp/mask, LDS re-stripe of e into the
// A-fragment layout, rowsum via WMMA vs all-ones B, O += e V (4 WMMA).
// d = 1024, dh = 64, H = 16 fixed.
// ---------------------------------------------------------------------------
__global__ __launch_bounds__(32) void attn_kernel(
    const float* __restrict__ Qp, const float* __restrict__ Kp,
    const float* __restrict__ Vp, const unsigned char* __restrict__ mask,
    float* __restrict__ Out, int Sq, int Sk, float scale) {
  constexpr int Dm = 1024;
  __shared__ __align__(64) __bf16 elds[16][32];

  const int lane = threadIdx.x & 31;
  const int m0 = lane & 15;
  const int kh = lane >> 4;
  const int b = blockIdx.z, h = blockIdx.y;
  const int q0 = blockIdx.x * 16;

  const float* Qb = Qp + (size_t)b * Sq * Dm + h * 64;
  const float* Kb = Kp + (size_t)b * Sk * Dm + h * 64;
  const float* Vb = Vp + (size_t)b * Sk * Dm + h * 64;

  // Q A-fragments: 16x64 -> two 16x32 k-steps (32B-aligned float4 loads)
  bf16x16 aq[2];
#pragma unroll
  for (int s = 0; s < 2; s++) {
    const float* qrow = Qb + (size_t)(q0 + m0) * Dm + s * 32 + kh * 16;
    float4 l0 = ((const float4*)qrow)[0];
    float4 l1 = ((const float4*)qrow)[1];
    float4 h0 = ((const float4*)(qrow + 8))[0];
    float4 h1 = ((const float4*)(qrow + 8))[1];
    aq[s][0]  = (__bf16)l0.x; aq[s][1]  = (__bf16)l0.y; aq[s][2]  = (__bf16)l0.z; aq[s][3]  = (__bf16)l0.w;
    aq[s][4]  = (__bf16)l1.x; aq[s][5]  = (__bf16)l1.y; aq[s][6]  = (__bf16)l1.z; aq[s][7]  = (__bf16)l1.w;
    aq[s][8]  = (__bf16)h0.x; aq[s][9]  = (__bf16)h0.y; aq[s][10] = (__bf16)h0.z; aq[s][11] = (__bf16)h0.w;
    aq[s][12] = (__bf16)h1.x; aq[s][13] = (__bf16)h1.y; aq[s][14] = (__bf16)h1.z; aq[s][15] = (__bf16)h1.w;
  }

  bf16x16 ones;
#pragma unroll
  for (int e = 0; e < 16; e++) ones[e] = (__bf16)1.0f;

  float maskrow[8];
#pragma unroll
  for (int r = 0; r < 8; r++) {
    int row = q0 + r + kh * 8;
    maskrow[r] = mask ? (mask[(size_t)b * Sq + row] ? 1.0f : 0.0f) : 1.0f;
  }

  f32x8 o[4];
#pragma unroll
  for (int t = 0; t < 4; t++) o[t] = {};
  f32x8 rs = {};

  for (int kt = 0; kt < Sk; kt += 32) {
    // ---- scores: two 16x16 tiles over a 32-key strip
    f32x8 sc[2];
#pragma unroll
    for (int j = 0; j < 2; j++) {
      f32x8 s = {};
#pragma unroll
      for (int st = 0; st < 2; st++) {
        const float* krow =
            Kb + (size_t)(kt + j * 16 + m0) * Dm + st * 32 + kh * 16;
        float4 k0v = ((const float4*)krow)[0];
        float4 k1v = ((const float4*)krow)[1];
        float4 k2v = ((const float4*)krow)[2];
        float4 k3v = ((const float4*)krow)[3];
        bf16x16 bk;
        bk[0]  = (__bf16)k0v.x; bk[1]  = (__bf16)k0v.y; bk[2]  = (__bf16)k0v.z; bk[3]  = (__bf16)k0v.w;
        bk[4]  = (__bf16)k1v.x; bk[5]  = (__bf16)k1v.y; bk[6]  = (__bf16)k1v.z; bk[7]  = (__bf16)k1v.w;
        bk[8]  = (__bf16)k2v.x; bk[9]  = (__bf16)k2v.y; bk[10] = (__bf16)k2v.z; bk[11] = (__bf16)k2v.w;
        bk[12] = (__bf16)k3v.x; bk[13] = (__bf16)k3v.y; bk[14] = (__bf16)k3v.z; bk[15] = (__bf16)k3v.w;
        s = WMMA_BF16(aq[st], bk, s);
      }
      sc[j] = s;
    }

    // ---- exp(clip(scale*s)) * mask  -> stage to LDS as bf16
#pragma unroll
    for (int j = 0; j < 2; j++)
#pragma unroll
      for (int r = 0; r < 8; r++) {
        float x = sc[j][r] * scale;
        x = fminf(fmaxf(x, -50.0f), 50.0f);
        float e = __expf(x) * maskrow[r];
        elds[r + kh * 8][j * 16 + m0] = (__bf16)e;
      }
    __syncthreads();  // single-wave block: compiler fence + LDS ordering

    // ---- reload e as A-fragment (16x32)
    bf16x16 ea;
    {
      const __bf16* ap = &elds[m0][0];
      ea = *(const bf16x16*)(ap + kh * 16);
    }

    // ---- row sums via WMMA against all-ones B
    rs = WMMA_BF16(ea, ones, rs);

    // ---- O += e @ V   (V B-frags: K = key index in tile, N = dh column;
    //      each of the 16 loads is coalesced across lanes)
#pragma unroll
    for (int t = 0; t < 4; t++) {
      bf16x16 bv;
#pragma unroll
      for (int e = 0; e < 16; e++)
        bv[e] = (__bf16)Vb[(size_t)(kt + kh * 16 + e) * Dm + t * 16 + m0];
      o[t] = WMMA_BF16(ea, bv, o[t]);
    }
    __syncthreads();
  }

  // ---- write out: out = O / max(sum, safe-1)
#pragma unroll
  for (int r = 0; r < 8; r++) {
    float s = rs[r];
    float denom = (s <= 0.0f) ? 1.0f : s;
    int row = q0 + r + kh * 8;
#pragma unroll
    for (int t = 0; t < 4; t++)
      Out[(size_t)b * Sq * Dm + (size_t)row * Dm + h * 64 + t * 16 + m0] =
          o[t][r] / denom;
  }
}

// ---------------------------------------------------------------------------
// Fused residual add + LayerNorm over d=1024. One 256-thread block per row,
// float4 per thread. out may alias resid (reads before writes, barriers).
// ---------------------------------------------------------------------------
__global__ __launch_bounds__(256) void add_ln_kernel(
    const float* __restrict__ resid, const float* __restrict__ branch,
    const float* __restrict__ gamma, const float* __restrict__ beta,
    float* __restrict__ out) {
  constexpr int Dm = 1024;
  const int row = blockIdx.x;
  const int tid = threadIdx.x;
  const float4* a4 = (const float4*)(resid + (size_t)row * Dm);
  const float4* c4 = (const float4*)(branch + (size_t)row * Dm);
  const float4* g4 = (const float4*)gamma;
  const float4* b4 = (const float4*)beta;
  float4* o4 = (float4*)(out + (size_t)row * Dm);

  __shared__ float sdata[256];

  float4 v = a4[tid];
  float4 c = c4[tid];
  v.x += c.x; v.y += c.y; v.z += c.z; v.w += c.w;
  float sum = v.x + v.y + v.z + v.w;

  sdata[tid] = sum;
  __syncthreads();
  for (int s = 128; s > 0; s >>= 1) {
    if (tid < s) sdata[tid] += sdata[tid + s];
    __syncthreads();
  }
  const float mu = sdata[0] * (1.0f / Dm);
  __syncthreads();

  float dx = v.x - mu, dy = v.y - mu, dz = v.z - mu, dw = v.w - mu;
  sdata[tid] = dx * dx + dy * dy + dz * dz + dw * dw;
  __syncthreads();
  for (int s = 128; s > 0; s >>= 1) {
    if (tid < s) sdata[tid] += sdata[tid + s];
    __syncthreads();
  }
  const float inv = rsqrtf(sdata[0] * (1.0f / Dm) + 1e-5f);

  float4 g = g4[tid], bb = b4[tid], r;
  r.x = dx * inv * g.x + bb.x;
  r.y = dy * inv * g.y + bb.y;
  r.z = dz * inv * g.z + bb.z;
  r.w = dw * inv * g.w + bb.w;
  o4[tid] = r;
}

// ---------------------------------------------------------------------------
// Host orchestration
// ---------------------------------------------------------------------------
static inline void launch_gemm(const float* A, const float* W, const float* bias,
                               float* C, int M, int N, int K, int act,
                               hipStream_t s) {
  dim3 grid(N / BN, M / BM);
  if (N == 1024 && K == 1024)
    gemm_bf16_kernel<1024, 1024><<<grid, 256, 0, s>>>(A, W, bias, C, act);
  else if (N == 4096 && K == 1024)
    gemm_bf16_kernel<4096, 1024><<<grid, 256, 0, s>>>(A, W, bias, C, act);
  else  // N == 1024, K == 4096
    gemm_bf16_kernel<1024, 4096><<<grid, 256, 0, s>>>(A, W, bias, C, act);
}

extern "C" void kernel_launch(void* const* d_in, const int* in_sizes, int n_in,
                              void* d_out, int out_size, void* d_ws, size_t ws_size,
                              hipStream_t stream) {
  (void)in_sizes; (void)n_in; (void)out_size; (void)ws_size;
  constexpr int Bsz = 4, SE = 1024, SD = 512, Dm = 1024, DFF = 4096, Hh = 16;
  constexpr int LE = 4, LD = 4;
  constexpr float SCALE = 0.125f;  // 1/sqrt(64)

  const float* cond_seq = (const float*)d_in[0];
  const float* main_seq = (const float*)d_in[1];
  const unsigned char* enc_mask = (const unsigned char*)d_in[2];
  const unsigned char* dec_mask = (const unsigned char*)d_in[3];
  const float* enc_attn_w = (const float*)d_in[4];
  const float* enc_ffn_w1 = (const float*)d_in[5];
  const float* enc_ffn_b1 = (const float*)d_in[6];
  const float* enc_ffn_w2 = (const float*)d_in[7];
  const float* enc_ffn_b2 = (const float*)d_in[8];
  const float* enc_ln_g   = (const float*)d_in[9];
  const float* enc_ln_b   = (const float*)d_in[10];
  const float* dec_attn_w = (const float*)d_in[11];
  const float* dec_ffn_w1 = (const float*)d_in[12];
  const float* dec_ffn_b1 = (const float*)d_in[13];
  const float* dec_ffn_w2 = (const float*)d_in[14];
  const float* dec_ffn_b2 = (const float*)d_in[15];
  const float* dec_ln_g   = (const float*)d_in[16];
  const float* dec_ln_b   = (const float*)d_in[17];

  float* ws = (float*)d_ws;
  size_t off = 0;
  float* X  = ws + off; off += (size_t)Bsz * SE * Dm;   // encoder stream / enc_tokens
  float* Yv = ws + off; off += (size_t)Bsz * SD * Dm;   // decoder stream
  float* Qb = ws + off; off += (size_t)Bsz * SE * Dm;
  float* Kb = ws + off; off += (size_t)Bsz * SE * Dm;
  float* Vb = ws + off; off += (size_t)Bsz * SE * Dm;
  float* T0 = ws + off; off += (size_t)Bsz * SE * Dm;   // attn output
  float* T2 = ws + off; off += (size_t)Bsz * SE * Dm;   // branch output
  float* T1 = ws + off; off += (size_t)Bsz * SE * DFF;  // ffn hidden

  hipMemcpyAsync(X, cond_seq, (size_t)Bsz * SE * Dm * sizeof(float),
                 hipMemcpyDeviceToDevice, stream);
  hipMemcpyAsync(Yv, main_seq, (size_t)Bsz * SD * Dm * sizeof(float),
                 hipMemcpyDeviceToDevice, stream);

  const int Me = Bsz * SE, Md = Bsz * SD;

  // ----- encoder -----
  for (int l = 0; l < LE; l++) {
    const float* aw = enc_attn_w + (size_t)l * 4 * Dm * Dm;
    launch_gemm(X, aw + 0 * (size_t)Dm * Dm, nullptr, Qb, Me, Dm, Dm, 0, stream);
    launch_gemm(X, aw + 1 * (size_t)Dm * Dm, nullptr, Kb, Me, Dm, Dm, 0, stream);
    launch_gemm(X, aw + 2 * (size_t)Dm * Dm, nullptr, Vb, Me, Dm, Dm, 0, stream);
    attn_kernel<<<dim3(SE / 16, Hh, Bsz), 32, 0, stream>>>(Qb, Kb, Vb, enc_mask,
                                                           T0, SE, SE, SCALE);
    launch_gemm(T0, aw + 3 * (size_t)Dm * Dm, nullptr, T2, Me, Dm, Dm, 0, stream);
    add_ln_kernel<<<Me, 256, 0, stream>>>(X, T2, enc_ln_g + ((size_t)l * 2 + 0) * Dm,
                                          enc_ln_b + ((size_t)l * 2 + 0) * Dm, X);
    launch_gemm(X, enc_ffn_w1 + (size_t)l * Dm * DFF, enc_ffn_b1 + (size_t)l * DFF,
                T1, Me, DFF, Dm, 1, stream);
    launch_gemm(T1, enc_ffn_w2 + (size_t)l * DFF * Dm, enc_ffn_b2 + (size_t)l * Dm,
                T2, Me, Dm, DFF, 0, stream);
    add_ln_kernel<<<Me, 256, 0, stream>>>(X, T2, enc_ln_g + ((size_t)l * 2 + 1) * Dm,
                                          enc_ln_b + ((size_t)l * 2 + 1) * Dm, X);
  }

  // ----- decoder -----
  for (int l = 0; l < LD; l++) {
    const float* aw = dec_attn_w + (size_t)l * 8 * Dm * Dm;
    // self-attention
    launch_gemm(Yv, aw + 0 * (size_t)Dm * Dm, nullptr, Qb, Md, Dm, Dm, 0, stream);
    launch_gemm(Yv, aw + 1 * (size_t)Dm * Dm, nullptr, Kb, Md, Dm, Dm, 0, stream);
    launch_gemm(Yv, aw + 2 * (size_t)Dm * Dm, nullptr, Vb, Md, Dm, Dm, 0, stream);
    attn_kernel<<<dim3(SD / 16, Hh, Bsz), 32, 0, stream>>>(Qb, Kb, Vb, dec_mask,
                                                           T0, SD, SD, SCALE);
    launch_gemm(T0, aw + 3 * (size_t)Dm * Dm, nullptr, T2, Md, Dm, Dm, 0, stream);
    add_ln_kernel<<<Md, 256, 0, stream>>>(Yv, T2, dec_ln_g + ((size_t)l * 3 + 0) * Dm,
                                          dec_ln_b + ((size_t)l * 3 + 0) * Dm, Yv);
    // cross-attention (K/V from encoder tokens, no mask)
    launch_gemm(Yv, aw + 4 * (size_t)Dm * Dm, nullptr, Qb, Md, Dm, Dm, 0, stream);
    launch_gemm(X,  aw + 5 * (size_t)Dm * Dm, nullptr, Kb, Me, Dm, Dm, 0, stream);
    launch_gemm(X,  aw + 6 * (size_t)Dm * Dm, nullptr, Vb, Me, Dm, Dm, 0, stream);
    attn_kernel<<<dim3(SD / 16, Hh, Bsz), 32, 0, stream>>>(Qb, Kb, Vb, nullptr,
                                                           T0, SD, SE, SCALE);
    launch_gemm(T0, aw + 7 * (size_t)Dm * Dm, nullptr, T2, Md, Dm, Dm, 0, stream);
    add_ln_kernel<<<Md, 256, 0, stream>>>(Yv, T2, dec_ln_g + ((size_t)l * 3 + 1) * Dm,
                                          dec_ln_b + ((size_t)l * 3 + 1) * Dm, Yv);
    // ffn
    launch_gemm(Yv, dec_ffn_w1 + (size_t)l * Dm * DFF, dec_ffn_b1 + (size_t)l * DFF,
                T1, Md, DFF, Dm, 1, stream);
    launch_gemm(T1, dec_ffn_w2 + (size_t)l * DFF * Dm, dec_ffn_b2 + (size_t)l * Dm,
                T2, Md, Dm, DFF, 0, stream);
    add_ln_kernel<<<Md, 256, 0, stream>>>(Yv, T2, dec_ln_g + ((size_t)l * 3 + 2) * Dm,
                                          dec_ln_b + ((size_t)l * 3 + 2) * Dm, Yv);
  }

  hipMemcpyAsync(d_out, Yv, (size_t)Bsz * SD * Dm * sizeof(float),
                 hipMemcpyDeviceToDevice, stream);
}
